// GAT_87222195848276
// MI455X (gfx1250) — matrix operand order repaired and checked by reading, not compiled, not run on hardware
//
#include <hip/hip_runtime.h>
#include <math.h>

#define N_NODES 100000
#define E_INTRA_N 1200000
#define E_INTER_N 400000
#define NGRAPH 256
#define DIM 64
#define GAT_SLOPE 0.2f
#define FC_SLOPE 0.01f
#define BN_EPS 1e-5f

typedef __attribute__((ext_vector_type(16))) _Float16 v16h;
typedef __attribute__((ext_vector_type(8)))  float    v8f;

// ---------------------------------------------------------------------------
// Float atomic max via signed-max / unsigned-min bit trick (monotone mapping).
// ---------------------------------------------------------------------------
__device__ inline void atomic_max_float(float* addr, float val) {
  if (val >= 0.0f)
    atomicMax((int*)addr, __float_as_int(val));
  else
    atomicMin((unsigned int*)addr, __float_as_uint(val));
}

// ---------------------------------------------------------------------------
// Generic Y = act(X @ W^T + bias) * (pscale*sfac) + pshift, D=64.
// One wave32 per 16-row tile; 8x v_wmma_f32_16x16x32_f16 per wave.
// act: 0=none, 1=silu, 2=relu, 3=leaky(slope)
// ---------------------------------------------------------------------------
__global__ __launch_bounds__(256)
void gemm64_act(const float* __restrict__ X, const float* __restrict__ W,
                const float* __restrict__ bias, float* __restrict__ Y,
                int nrows, int act, float slope,
                const float* __restrict__ pscale, const float* __restrict__ pshift,
                float sfac)
{
  const int wave = threadIdx.x >> 5;
  const int lane = threadIdx.x & 31;
  const int row0 = (blockIdx.x * 8 + wave) * 16;
  if (row0 >= nrows) return;            // wave-uniform: EXEC stays all-ones
  const int ml = lane & 15;
  const int hsel = lane >> 4;

  v8f acc[4];
  #pragma unroll
  for (int t = 0; t < 4; ++t) acc[t] = (v8f)0.0f;

  #pragma unroll
  for (int kt = 0; kt < 2; ++kt) {
    const int k0 = kt * 32;
    // --- A fragment: 16x32 f16, ISA layout (lane=M, halves split K) ---
    v16h a;
    {
      const float* xr = X + (size_t)(row0 + ml) * DIM + k0 + hsel * 8;
      #pragma unroll
      for (int j = 0; j < 4; ++j) {
        a[2*j]     = (_Float16)xr[2*j];
        a[2*j + 1] = (_Float16)xr[2*j + 1];
        a[2*j + 8] = (_Float16)xr[16 + 2*j];
        a[2*j + 9] = (_Float16)xr[16 + 2*j + 1];
      }
    }
    // --- B fragments: 32x16 f16, B[k][n] = W[n][k] ---
    #pragma unroll
    for (int nt = 0; nt < 4; ++nt) {
      v16h b;
      const float* wr = W + (size_t)(nt * 16 + ml) * DIM + k0 + hsel * 8;
      #pragma unroll
      for (int j = 0; j < 4; ++j) {
        b[2*j]     = (_Float16)wr[2*j];
        b[2*j + 1] = (_Float16)wr[2*j + 1];
        b[2*j + 8] = (_Float16)wr[16 + 2*j];
        b[2*j + 9] = (_Float16)wr[16 + 2*j + 1];
      }
      acc[nt] = __builtin_amdgcn_wmma_f32_16x16x32_f16(
          false, a, false, b, (short)0, acc[nt], false, false);
    }
  }

  // --- epilogue: bias + activation + optional BN affine ---
  #pragma unroll
  for (int nt = 0; nt < 4; ++nt) {
    const int n = nt * 16 + ml;
    const float bn = bias ? bias[n] : 0.0f;
    const float ps = pscale ? pscale[n] * sfac : 1.0f;
    const float ph = pshift ? pshift[n] : 0.0f;
    #pragma unroll
    for (int r = 0; r < 8; ++r) {
      const int m = r + hsel * 8;
      float v = acc[nt][r] + bn;
      if (act == 1)      v = v / (1.0f + __expf(-v));   // silu
      else if (act == 2) v = fmaxf(v, 0.0f);            // relu
      else if (act == 3) v = (v > 0.0f) ? v : v * slope;
      v = v * ps + ph;
      Y[(size_t)(row0 + m) * DIM + n] = v;
    }
  }
}

// ---------------------------------------------------------------------------
// Per-node attention scalars: a_s[i] = h'[i].att_src, a_d[i] = h'[i].att_dst
// ---------------------------------------------------------------------------
__global__ void node_dots(const float* __restrict__ H,
                          const float* __restrict__ as_w,
                          const float* __restrict__ ad_w,
                          float* __restrict__ a_s, float* __restrict__ a_d, int n)
{
  int i = blockIdx.x * blockDim.x + threadIdx.x;
  if (i >= n) return;
  const float* h = H + (size_t)i * DIM;
  float s = 0.0f, d = 0.0f;
  #pragma unroll 8
  for (int k = 0; k < DIM; ++k) { float v = h[k]; s += v * as_w[k]; d += v * ad_w[k]; }
  a_s[i] = s; a_d[i] = d;
}

// ---------------------------------------------------------------------------
// Edge endpoint decode: [intra | inter | self-loops]
// edge_index layout is (2,E) row-major: [0..E)=src, [E..2E)=dst
// ---------------------------------------------------------------------------
__device__ inline void edge_endpoints(int e, const int* __restrict__ ia,
                                      const int* __restrict__ ib, int& s, int& d)
{
  if (e < E_INTRA_N)                     { s = ia[e]; d = ia[E_INTRA_N + e]; }
  else if (e < E_INTRA_N + E_INTER_N)    { int t = e - E_INTRA_N; s = ib[t]; d = ib[E_INTER_N + t]; }
  else                                   { s = d = e - (E_INTRA_N + E_INTER_N); }
}

__global__ void edge_logits(const int* __restrict__ ia, const int* __restrict__ ib,
                            const float* __restrict__ a_s, const float* __restrict__ a_d,
                            float* __restrict__ ebuf, float* __restrict__ emax, int etot)
{
  int e = blockIdx.x * blockDim.x + threadIdx.x;
  if (e >= etot) return;
  int s, d; edge_endpoints(e, ia, ib, s, d);
  float v = a_s[s] + a_d[d];
  v = (v > 0.0f) ? v : v * GAT_SLOPE;
  ebuf[e] = v;
  atomic_max_float(&emax[d], v);
}

__global__ void edge_exp(const int* __restrict__ ia, const int* __restrict__ ib,
                         float* __restrict__ ebuf, const float* __restrict__ emax,
                         float* __restrict__ denom, int etot)
{
  int e = blockIdx.x * blockDim.x + threadIdx.x;
  if (e >= etot) return;
  int s, d; edge_endpoints(e, ia, ib, s, d);
  float ex = __expf(ebuf[e] - emax[d]);
  ebuf[e] = ex;
  atomicAdd(&denom[d], ex);
}

__global__ void edge_scatter(const int* __restrict__ ia, const int* __restrict__ ib,
                             const float* __restrict__ ebuf, const float* __restrict__ denom,
                             const float* __restrict__ Hp, float* __restrict__ acc, int etot)
{
  int e = blockIdx.x * blockDim.x + threadIdx.x;
  if (e >= etot) return;
  int s, d; edge_endpoints(e, ia, ib, s, d);
  float alpha = ebuf[e] / denom[d];
  const float4* hp = (const float4*)(Hp + (size_t)s * DIM);
  float* o = acc + (size_t)d * DIM;
  #pragma unroll
  for (int q = 0; q < DIM / 4; ++q) {
    float4 v = hp[q];
    atomicAdd(o + 4*q + 0, alpha * v.x);
    atomicAdd(o + 4*q + 1, alpha * v.y);
    atomicAdd(o + 4*q + 2, alpha * v.z);
    atomicAdd(o + 4*q + 3, alpha * v.w);
  }
}

__global__ void bias_act(const float* __restrict__ acc, const float* __restrict__ bias,
                         float* __restrict__ out, int total, int relu)
{
  int i = blockIdx.x * blockDim.x + threadIdx.x;
  if (i >= total) return;
  float v = acc[i] + bias[i & (DIM - 1)];
  if (relu) v = fmaxf(v, 0.0f);
  out[i] = v;
}

__global__ void pool_sum(const float* __restrict__ H, const int* __restrict__ batch,
                         float* __restrict__ g, int n)
{
  int i = blockIdx.x * blockDim.x + threadIdx.x;
  if (i >= n) return;
  int b = batch[i];
  const float4* h = (const float4*)(H + (size_t)i * DIM);
  float* o = g + (size_t)b * DIM;
  #pragma unroll
  for (int q = 0; q < DIM / 4; ++q) {
    float4 v = h[q];
    atomicAdd(o + 4*q + 0, v.x);
    atomicAdd(o + 4*q + 1, v.y);
    atomicAdd(o + 4*q + 2, v.z);
    atomicAdd(o + 4*q + 3, v.w);
  }
}

__global__ void fc3_kernel(const float* __restrict__ G, const float* __restrict__ w,
                           const float* __restrict__ b, float* __restrict__ out)
{
  int i = blockIdx.x * blockDim.x + threadIdx.x;
  if (i >= NGRAPH) return;
  const float* r = G + (size_t)i * DIM;
  float s = 0.0f;
  #pragma unroll 8
  for (int k = 0; k < DIM; ++k) s += r[k] * w[k];
  out[i] = s + b[0];
}

__global__ void fill_kernel(float* __restrict__ p, float v, int cnt)
{
  int i = blockIdx.x * blockDim.x + threadIdx.x;
  if (i < cnt) p[i] = v;
}

// ---------------------------------------------------------------------------
extern "C" void kernel_launch(void* const* d_in, const int* in_sizes, int n_in,
                              void* d_out, int out_size, void* d_ws, size_t ws_size,
                              hipStream_t stream)
{
  const float* x      = (const float*)d_in[0];
  const int*   ei_a   = (const int*)d_in[1];
  const int*   ei_b   = (const int*)d_in[2];
  const int*   batch  = (const int*)d_in[3];
  const float* lin1_w = (const float*)d_in[4];
  const float* lin1_b = (const float*)d_in[5];
  const float* gw[3]  = {(const float*)d_in[6],  (const float*)d_in[10], (const float*)d_in[14]};
  const float* gas[3] = {(const float*)d_in[7],  (const float*)d_in[11], (const float*)d_in[15]};
  const float* gad[3] = {(const float*)d_in[8],  (const float*)d_in[12], (const float*)d_in[16]};
  const float* gb[3]  = {(const float*)d_in[9],  (const float*)d_in[13], (const float*)d_in[17]};
  const float* fc1_w  = (const float*)d_in[18];
  const float* fc1_b  = (const float*)d_in[19];
  const float* bn1_g  = (const float*)d_in[20];
  const float* bn1_b  = (const float*)d_in[21];
  const float* fc2_w  = (const float*)d_in[22];
  const float* fc2_b  = (const float*)d_in[23];
  const float* bn2_g  = (const float*)d_in[24];
  const float* bn2_b  = (const float*)d_in[25];
  const float* fc3_w  = (const float*)d_in[26];
  const float* fc3_b  = (const float*)d_in[27];

  const size_t N64 = (size_t)N_NODES * DIM;
  const int ETOT = E_INTRA_N + E_INTER_N + N_NODES;

  float* ws    = (float*)d_ws;
  float* hcur  = ws;                 // N x 64
  float* hpr   = hcur  + N64;        // N x 64
  float* hacc  = hpr   + N64;        // N x 64
  float* a_s   = hacc  + N64;        // N
  float* a_d   = a_s   + N_NODES;    // N
  float* emax  = a_d   + N_NODES;    // N
  float* denom = emax  + N_NODES;    // N
  float* ebuf  = denom + N_NODES;    // ETOT
  float* g0    = ebuf  + ETOT;       // 256 x 64
  float* g1    = g0    + NGRAPH*DIM;
  float* g2    = g1    + NGRAPH*DIM;

  const float inv_bn = 1.0f / sqrtf(1.0f + BN_EPS);

  dim3 blk(256);
  const int grid_gemm = (N_NODES + 127) / 128;   // 128 rows per 8-wave block
  const int grid_n    = (N_NODES + 255) / 256;
  const int grid_n64  = (int)((N64 + 255) / 256);
  const int grid_e    = (ETOT + 255) / 256;
  const int grid_g64  = (NGRAPH * DIM + 255) / 256;

  // lin1 + SiLU
  gemm64_act<<<grid_gemm, blk, 0, stream>>>(x, lin1_w, lin1_b, hcur, N_NODES,
                                            1, 0.0f, nullptr, nullptr, 1.0f);

  for (int L = 0; L < 3; ++L) {
    // h' = h @ W^T (bias applied post-aggregation)
    gemm64_act<<<grid_gemm, blk, 0, stream>>>(hcur, gw[L], nullptr, hpr, N_NODES,
                                              0, 0.0f, nullptr, nullptr, 1.0f);
    node_dots<<<grid_n, blk, 0, stream>>>(hpr, gas[L], gad[L], a_s, a_d, N_NODES);

    fill_kernel<<<grid_n,   blk, 0, stream>>>(emax, -INFINITY, N_NODES);
    fill_kernel<<<grid_n,   blk, 0, stream>>>(denom, 0.0f, N_NODES);
    fill_kernel<<<grid_n64, blk, 0, stream>>>(hacc, 0.0f, (int)N64);

    edge_logits <<<grid_e, blk, 0, stream>>>(ei_a, ei_b, a_s, a_d, ebuf, emax, ETOT);
    edge_exp    <<<grid_e, blk, 0, stream>>>(ei_a, ei_b, ebuf, emax, denom, ETOT);
    edge_scatter<<<grid_e, blk, 0, stream>>>(ei_a, ei_b, ebuf, denom, hpr, hacc, ETOT);

    bias_act<<<grid_n64, blk, 0, stream>>>(hacc, gb[L], hcur, (int)N64, (L < 2) ? 1 : 0);
  }

  // global_add_pool
  fill_kernel<<<grid_g64, blk, 0, stream>>>(g0, 0.0f, NGRAPH * DIM);
  pool_sum<<<grid_n, blk, 0, stream>>>(hcur, batch, g0, N_NODES);

  // FC head: leaky + BN affine folded into GEMM epilogue
  gemm64_act<<<(NGRAPH + 127) / 128, blk, 0, stream>>>(g0, fc1_w, fc1_b, g1, NGRAPH,
                                                       3, FC_SLOPE, bn1_g, bn1_b, inv_bn);
  gemm64_act<<<(NGRAPH + 127) / 128, blk, 0, stream>>>(g1, fc2_w, fc2_b, g2, NGRAPH,
                                                       3, FC_SLOPE, bn2_g, bn2_b, inv_bn);
  fc3_kernel<<<1, 256, 0, stream>>>(g2, fc3_w, fc3_b, (float*)d_out);
}
